// _Decoder_80676665688692
// MI455X (gfx1250) — matrix operand order repaired and checked by reading, not compiled, not run on hardware
//
#include <hip/hip_runtime.h>
#include <hip/hip_bf16.h>

typedef __attribute__((ext_vector_type(16))) _Float16 v16h;
typedef __attribute__((ext_vector_type(8)))  _Float16 v8h;
typedef __attribute__((ext_vector_type(8)))  float    v8f;

namespace {
constexpr int BB = 8, TT = 50, NN = 50, FF = 8, HH = 64, KK = 4;
constexpr int EE     = NN * (NN - 1);      // 2450 edges
constexpr int ETILES = (EE + 15) / 16;     // 154 row tiles of 16 edges
constexpr int ASTR   = 104;                // act row stride in halves (16B multiple, bank-friendly)
constexpr int AGGS   = 72;                 // agg row stride in floats (bank-friendly)
constexpr int WREG   = 11264;              // halves: max(msg 10240, out 11264) weight region
}

__device__ __forceinline__ float elu_f(float v) {
    return v > 0.f ? v : (__expf(v) - 1.f);
}

__device__ __forceinline__ v8f wmma16(v16h a, v16h b, v8f c) {
    // D(f32 16x16) = A(f16 16x32) * B(f16 32x16) + C
    return __builtin_amdgcn_wmma_f32_16x16x32_f16(false, a, false, b, (short)0, c, false, false);
}

// A-fragment for 16-bit 16x32 WMMA, K-chunk c, from an LDS activation row.
// lanes 0-15: elems 0..7 = K(32c+0..7),  8..15 = K(32c+16..23)
// lanes16-31: elems 0..7 = K(32c+8..15), 8..15 = K(32c+24..31)
__device__ __forceinline__ v16h load_a_chunk(const _Float16* row, int c, int kh) {
    const _Float16* p = row + c * 32 + kh * 8;
    v8h lo = *(const v8h*)p;
    v8h hi = *(const v8h*)(p + 16);
    v16h a;
#pragma unroll
    for (int k = 0; k < 8; ++k) { a[k] = lo[k]; a[k + 8] = hi[k]; }
    return a;
}

__global__ void __launch_bounds__(256)
nri_decoder(const float* __restrict__ x, const float* __restrict__ edges,
            const float* __restrict__ mW1, const float* __restrict__ mb1,
            const float* __restrict__ mW2, const float* __restrict__ mb2,
            const float* __restrict__ mW3, const float* __restrict__ mb3,
            const float* __restrict__ oW1, const float* __restrict__ ob1,
            const float* __restrict__ oW2, const float* __restrict__ ob2,
            const float* __restrict__ oW3, const float* __restrict__ ob3,
            float* __restrict__ out)
{
    // LDS budget: 22528 + 26624 + 14400 + 768 = 64320 B  (<= 64KB)
    __shared__ __align__(32) _Float16 sW[WREG];           // transposed f16 weights Wt[h][k]
    __shared__ __align__(32) _Float16 sAct[8 * 16 * ASTR];// per-wave 16-row activation scratch
    __shared__ float sAgg[NN * AGGS];                     // per-(b,t) node aggregation [50][64+pad]
    __shared__ float sB1[HH], sB2[HH], sB3[HH];

    const int tid  = threadIdx.x;
    const int lane = tid & 31;
    const int wave = tid >> 5;
    const int m    = lane & 15;   // row (A) or column (B/C/D) index within a 16-tile
    const int kh   = lane >> 4;   // lane-half selector for K ranges / M halves
    const int bt   = blockIdx.x;  // one block per (b,t)
    const int b    = bt / TT;

    const float* xbt   = x + (size_t)bt * NN * FF;
    _Float16*    myact = sAct + wave * 16 * ASTR;

    for (int i = tid; i < NN * AGGS; i += 256) sAgg[i] = 0.f;

    // ======================= edge phase =======================
    for (int ty = 1; ty < KK; ++ty) {           // edge type 0 skipped, as in reference
        __syncthreads();                        // previous users of sW done
        // wt1 @0 : [64][32] (K=16 zero-padded to 32); wt2 @2048, wt3 @6144 : [64][64]
        for (int i = tid; i < 64 * 32; i += 256) {
            int h = i >> 5, k = i & 31;
            sW[i] = (k < 16) ? (_Float16)mW1[(ty * 16 + k) * 64 + h] : (_Float16)0.f;
        }
        for (int i = tid; i < 64 * 64; i += 256) {
            int h = i >> 6, k = i & 63;
            sW[2048 + i] = (_Float16)mW2[(ty * 64 + k) * 64 + h];
            sW[6144 + i] = (_Float16)mW3[(ty * 64 + k) * 64 + h];
        }
        if (tid < 64) {
            sB1[tid] = mb1[ty * 64 + tid];
            sB2[tid] = mb2[ty * 64 + tid];
            sB3[tid] = mb3[ty * 64 + tid];
        }
        __syncthreads();

        for (int tile = wave; tile < ETILES; tile += 8) {
            const int ebase = tile * 16;

            // ---- layer 1: A built directly from x (send || recv), K=16 pad 32 ----
            int e0 = ebase + m;
            int ec = e0 < EE ? e0 : EE - 1;              // clamp; invalid rows get weight 0 later
            int r0 = ec / (NN - 1);
            int j0 = ec - r0 * (NN - 1);
            int s0 = j0 + (j0 >= r0 ? 1 : 0);
            const float* np = xbt + (kh ? r0 : s0) * FF; // K 0..7 = sender, 8..15 = receiver
            v16h a;
#pragma unroll
            for (int k = 0; k < 8; ++k) a[k] = (_Float16)np[k];
#pragma unroll
            for (int k = 8; k < 16; ++k) a[k] = (_Float16)0.f;

            v8f c1[4] = {};
#pragma unroll
            for (int nt = 0; nt < 4; ++nt) {
                v16h bw = load_a_chunk(&sW[(nt * 16 + m) * 32], 0, kh);
                c1[nt] = wmma16(a, bw, c1[nt]);
            }
#pragma unroll
            for (int nt = 0; nt < 4; ++nt) {
                int h = nt * 16 + m;
                float bias = sB1[h];
#pragma unroll
                for (int j = 0; j < 8; ++j)
                    myact[(kh * 8 + j) * ASTR + h] = (_Float16)elu_f(c1[nt][j] + bias);
            }

            // ---- layer 2: K=64 ----
            v8f c2[4] = {};
#pragma unroll
            for (int c = 0; c < 2; ++c) {
                v16h a2 = load_a_chunk(myact + m * ASTR, c, kh);
#pragma unroll
                for (int nt = 0; nt < 4; ++nt) {
                    v16h bw = load_a_chunk(&sW[2048 + (nt * 16 + m) * 64], c, kh);
                    c2[nt] = wmma16(a2, bw, c2[nt]);
                }
            }
#pragma unroll
            for (int nt = 0; nt < 4; ++nt) {
                int h = nt * 16 + m;
                float bias = sB2[h];
#pragma unroll
                for (int j = 0; j < 8; ++j)
                    myact[(kh * 8 + j) * ASTR + h] = (_Float16)elu_f(c2[nt][j] + bias);
            }

            // ---- layer 3 + edge-weighted scatter-add to receivers ----
            v8f c3[4] = {};
#pragma unroll
            for (int c = 0; c < 2; ++c) {
                v16h a3 = load_a_chunk(myact + m * ASTR, c, kh);
#pragma unroll
                for (int nt = 0; nt < 4; ++nt) {
                    v16h bw = load_a_chunk(&sW[6144 + (nt * 16 + m) * 64], c, kh);
                    c3[nt] = wmma16(a3, bw, c3[nt]);
                }
            }
            float wrow[8]; int rrow[8];
#pragma unroll
            for (int j = 0; j < 8; ++j) {
                int e   = ebase + kh * 8 + j;     // row M = kh*8 + j of the C tile
                int ecl = e < EE ? e : EE - 1;
                rrow[j] = ecl / (NN - 1);
                wrow[j] = (e < EE) ? edges[((size_t)b * EE + e) * KK + ty] : 0.f;
            }
#pragma unroll
            for (int nt = 0; nt < 4; ++nt) {
                int h = nt * 16 + m;
                float bias = sB3[h];
#pragma unroll
                for (int j = 0; j < 8; ++j)
                    atomicAdd(&sAgg[rrow[j] * AGGS + h], (c3[nt][j] + bias) * wrow[j]);
            }
        }
    }

    // ======================= node phase =======================
    __syncthreads();   // all ds_add_f32 into sAgg visible; sW free to overwrite
    // wto1 @0 : [64][96] (K=72 pad 96); wto2 @6144 : [64][64]; wto3 @10240 : [16][64] (8 valid cols)
    for (int i = tid; i < 64 * 96; i += 256) {
        int h = i / 96, k = i - h * 96;
        sW[i] = (k < FF + HH) ? (_Float16)oW1[k * 64 + h] : (_Float16)0.f;
    }
    for (int i = tid; i < 64 * 64; i += 256) {
        int h = i >> 6, k = i & 63;
        sW[6144 + i] = (_Float16)oW2[k * 64 + h];
    }
    for (int i = tid; i < 16 * 64; i += 256) {
        int h = i >> 6, k = i & 63;
        sW[10240 + i] = (h < FF) ? (_Float16)oW3[k * FF + h] : (_Float16)0.f;
    }
    if (tid < 64) { sB1[tid] = ob1[tid]; sB2[tid] = ob2[tid]; }
    if (tid < 8)  sB3[tid] = ob3[tid];
    __syncthreads();

    if (wave < 4) {                   // 4 waves cover 50 nodes (tiles of 16, wave-uniform branch)
        int  n   = wave * 16 + m;
        int  ncl = n < NN ? n : NN - 1;
        bool nv  = n < NN;
        const float* xr = xbt + ncl * FF;
        for (int c = kh * 48; c < kh * 48 + 48; ++c) {   // build aug = [x | agg | 0pad]
            float v = 0.f;
            if (nv) {
                if (c < FF)            v = xr[c];
                else if (c < FF + HH)  v = sAgg[ncl * AGGS + (c - FF)];
            }
            myact[m * ASTR + c] = (_Float16)v;
        }

        // out layer 1: K=96 (72 real)
        v8f c1[4] = {};
#pragma unroll
        for (int c = 0; c < 3; ++c) {
            v16h a1 = load_a_chunk(myact + m * ASTR, c, kh);
#pragma unroll
            for (int nt = 0; nt < 4; ++nt) {
                v16h bw = load_a_chunk(&sW[(nt * 16 + m) * 96], c, kh);
                c1[nt] = wmma16(a1, bw, c1[nt]);
            }
        }
#pragma unroll
        for (int nt = 0; nt < 4; ++nt) {
            int h = nt * 16 + m;
            float bias = sB1[h];
#pragma unroll
            for (int j = 0; j < 8; ++j)
                myact[(kh * 8 + j) * ASTR + h] = (_Float16)elu_f(c1[nt][j] + bias);
        }

        // out layer 2: K=64
        v8f c2[4] = {};
#pragma unroll
        for (int c = 0; c < 2; ++c) {
            v16h a2 = load_a_chunk(myact + m * ASTR, c, kh);
#pragma unroll
            for (int nt = 0; nt < 4; ++nt) {
                v16h bw = load_a_chunk(&sW[6144 + (nt * 16 + m) * 64], c, kh);
                c2[nt] = wmma16(a2, bw, c2[nt]);
            }
        }
#pragma unroll
        for (int nt = 0; nt < 4; ++nt) {
            int h = nt * 16 + m;
            float bias = sB2[h];
#pragma unroll
            for (int j = 0; j < 8; ++j)
                myact[(kh * 8 + j) * ASTR + h] = (_Float16)elu_f(c2[nt][j] + bias);
        }

        // out layer 3: K=64, single N tile (8 valid output features), residual store
        v8f c3 = {};
#pragma unroll
        for (int c = 0; c < 2; ++c) {
            v16h a3 = load_a_chunk(myact + m * ASTR, c, kh);
            v16h bw = load_a_chunk(&sW[10240 + m * 64], c, kh);
            c3 = wmma16(a3, bw, c3);
        }
        if (m < FF) {
            float bias = sB3[m];
#pragma unroll
            for (int j = 0; j < 8; ++j) {
                int nn2 = wave * 16 + kh * 8 + j;   // row M of C tile = node in tile
                if (nn2 < NN) {
                    size_t o = (size_t)bt * NN * FF + (size_t)nn2 * FF + m;
                    out[o] = x[o] + c3[j] + bias;
                }
            }
        }
    }
}

extern "C" void kernel_launch(void* const* d_in, const int* in_sizes, int n_in,
                              void* d_out, int out_size, void* d_ws, size_t ws_size,
                              hipStream_t stream) {
    const float* x     = (const float*)d_in[0];
    const float* edges = (const float*)d_in[1];
    // d_in[2]=receiver, d_in[3]=sender: one-hot structure exploited analytically (e -> r,s)
    const float* mW1 = (const float*)d_in[4];
    const float* mb1 = (const float*)d_in[5];
    const float* mW2 = (const float*)d_in[6];
    const float* mb2 = (const float*)d_in[7];
    const float* mW3 = (const float*)d_in[8];
    const float* mb3 = (const float*)d_in[9];
    const float* oW1 = (const float*)d_in[10];
    const float* ob1 = (const float*)d_in[11];
    const float* oW2 = (const float*)d_in[12];
    const float* ob2 = (const float*)d_in[13];
    const float* oW3 = (const float*)d_in[14];
    const float* ob3 = (const float*)d_in[15];

    nri_decoder<<<BB * TT, 256, 0, stream>>>(x, edges,
                                             mW1, mb1, mW2, mb2, mW3, mb3,
                                             oW1, ob1, oW2, ob2, oW3, ob3,
                                             (float*)d_out);
}
